// MDDNet_20023137533996
// MI455X (gfx1250) — matrix operand order repaired and checked by reading, not compiled, not run on hardware
//
#include <hip/hip_runtime.h>
#include <hip/hip_bf16.h>

#define NN 20000
#define EE 320000
#define DD 128
#define H1 256
#define H2 512
#define H3 1024
#define SLOPEC 0.01f
#define EPSC 1e-5f

typedef __attribute__((ext_vector_type(16))) __bf16 v16bf;
typedef __attribute__((ext_vector_type(8)))  float  v8f;

union U16B { uint4 q[2]; v16bf v; };

static __device__ __forceinline__ v16bf make_frag(uint4 lo, uint4 hi) {
  U16B u; u.q[0] = lo; u.q[1] = hi; return u.v;
}

// A-matrix 16x32 bf16 fragment from LDS/global tile at (row0, k0), row stride ldk elems.
// Lane layout (ISA 7.12.2): lanes 0-15 row M=lane, K chunks [k0..k0+7] and [k0+16..k0+23];
// lanes 16-31 row M=lane-16, K chunks [k0+8..15] and [k0+24..31].
static __device__ __forceinline__ v16bf afrag(const __bf16* base, int ldk,
                                              int row0, int k0, int lh, int lr) {
  const char* p = (const char*)(base + (size_t)(row0 + lr) * ldk + k0 + lh * 8);
  uint4 lo = *(const uint4*)p;
  uint4 hi = *(const uint4*)(p + 32);
  return make_frag(lo, hi);
}

// B-matrix 32x16 bf16 fragment from transposed weights Wt[n][k] (row stride K elems).
// Lane layout: lanes 0-15 N=lane, K [k0..k0+7],[k0+16..k0+23]; lanes 16-31 N=lane-16, K [k0+8..15],[k0+24..31].
static __device__ __forceinline__ v16bf bfrag(const __bf16* Wt, int K,
                                              int n0, int k0, int lh, int ln) {
  const char* p = (const char*)(Wt + (size_t)(n0 + ln) * K + k0 + lh * 8);
  uint4 lo = *(const uint4*)p;
  uint4 hi = *(const uint4*)(p + 32);
  return make_frag(lo, hi);
}

static __device__ __forceinline__ v8f wmma_bf16(v16bf a, v16bf b, v8f c) {
  return __builtin_amdgcn_wmma_f32_16x16x32_bf16(false, a, false, b, (short)0, c,
                                                 false, false);
}

// Bias preloaded into the accumulator: C/D lane layout holds a single column
// (N = lane%16) across 8 rows, so a per-column bias is a plain splat.
static __device__ __forceinline__ v8f splat8(float b) {
  v8f z = {b, b, b, b, b, b, b, b};
  return z;
}

// leaky = max(x,0) + slope*min(x,0)  ->  v_max_num + v_min_num + v_fmac
static __device__ __forceinline__ float lrelu(float x) {
  return fmaf(SLOPEC, fminf(x, 0.f), fmaxf(x, 0.f));
}

static __device__ __forceinline__ unsigned pk2(float a, float b) {
  unsigned short ua = __builtin_bit_cast(unsigned short, (__bf16)a);
  unsigned short ub = __builtin_bit_cast(unsigned short, (__bf16)b);
  return (unsigned)ua | ((unsigned)ub << 16);
}

// ---------------------------------------------------------------------------
// Weight prep: W[k][n] (f32) -> Wt[n][k] (bf16), contiguous in k for B-frag loads.
__global__ void __launch_bounds__(256) cvt_transpose(const float* __restrict__ in,
                                                     __bf16* __restrict__ out,
                                                     int K, int Nout) {
  int idx = blockIdx.x * 256 + threadIdx.x;
  if (idx >= K * Nout) return;
  int n = idx / K, k = idx - n * K;
  out[idx] = (__bf16)in[(size_t)k * Nout + n];
}

// ---------------------------------------------------------------------------
// Edge MLP: per block 128 edges. A0 = bf16(x0[src]*edge_attr) in LDS,
// mm1 (128->256) + leaky -> LDS, mm2 (256->512) + leaky -> f32 atomic scatter.
// Wave blocking: 8 waves = 2 M-groups (64 rows each) x 4 N-groups; each wave
// computes 4 M-tiles per B-fragment load (4x less global weight traffic).
__global__ void __launch_bounds__(256) edge_mlp(const float* __restrict__ x0,
                                                const int* __restrict__ eidx,
                                                const float* __restrict__ eattr,
                                                const __bf16* __restrict__ W1t,
                                                const float* __restrict__ b1,
                                                const __bf16* __restrict__ W2t,
                                                const float* __restrict__ b2,
                                                float* __restrict__ aggr) {
  extern __shared__ char smem[];
  __bf16* A0   = (__bf16*)smem;                              // 128 x 128 bf16
  __bf16* ACT1 = (__bf16*)(smem + 128 * DD * 2);             // 128 x 256 bf16
  int*    DSTS = (int*)(smem + 128 * DD * 2 + 128 * H1 * 2); // 128 ints

  const int t = threadIdx.x;
  const int blk = blockIdx.x;

  { // stage A0 = x_j * edge_attr (bf16), and dst indices
    int r = t >> 1, half = t & 1;
    int e = blk * 128 + r;
    int src = eidx[e];
    if (!half) DSTS[r] = eidx[EE + e];
    const float4* xr = (const float4*)(x0 + (size_t)src * DD) + half * 16;
    const float4* ar = (const float4*)(eattr + (size_t)e * DD) + half * 16;
    uint2* orow = (uint2*)(A0 + r * DD + half * 64);
#pragma unroll
    for (int i = 0; i < 16; ++i) {
      float4 a = xr[i];
      float4 w = ar[i];
      uint2 pk;
      pk.x = pk2(a.x * w.x, a.y * w.y);
      pk.y = pk2(a.z * w.z, a.w * w.w);
      orow[i] = pk;
    }
  }
  __syncthreads();

  const int wave = t >> 5, lane = t & 31;
  const int lh = lane >> 4, ln = lane & 15;
  const int mg = wave & 1;   // M-group: rows mg*64 .. mg*64+63 (4 tiles)
  const int wg = wave >> 1;  // N-group 0..3
  const int mrow0 = mg * 64;

  // mm1: K = 128; this wave: 4 N-tiles (wg+4i), 4 M-tiles each
  for (int i = 0; i < 4; ++i) {
    int n0 = (wg + 4 * i) * 16;
    float bias = b1[n0 + ln];
    v8f acc0 = splat8(bias), acc1 = splat8(bias), acc2 = splat8(bias), acc3 = splat8(bias);
#pragma unroll
    for (int kk = 0; kk < 4; ++kk) {
      v16bf bf = bfrag(W1t, DD, n0, kk * 32, lh, ln);
      acc0 = wmma_bf16(afrag(A0, DD, mrow0 +  0, kk * 32, lh, ln), bf, acc0);
      acc1 = wmma_bf16(afrag(A0, DD, mrow0 + 16, kk * 32, lh, ln), bf, acc1);
      acc2 = wmma_bf16(afrag(A0, DD, mrow0 + 32, kk * 32, lh, ln), bf, acc2);
      acc3 = wmma_bf16(afrag(A0, DD, mrow0 + 48, kk * 32, lh, ln), bf, acc3);
    }
    v8f accs[4] = {acc0, acc1, acc2, acc3};
#pragma unroll
    for (int m = 0; m < 4; ++m)
#pragma unroll
      for (int v = 0; v < 8; ++v) {
        int row = mrow0 + m * 16 + v + lh * 8;
        ACT1[row * H1 + n0 + ln] = (__bf16)lrelu(accs[m][v]);
      }
  }
  __syncthreads();

  // mm2: K = 256; this wave: 8 N-tiles (wg+4i), 4 M-tiles each; scatter-add out
  for (int i = 0; i < 8; ++i) {
    int n0 = (wg + 4 * i) * 16;
    float bias = b2[n0 + ln];
    v8f acc0 = splat8(bias), acc1 = splat8(bias), acc2 = splat8(bias), acc3 = splat8(bias);
#pragma unroll
    for (int kk = 0; kk < 8; ++kk) {
      v16bf bf = bfrag(W2t, H1, n0, kk * 32, lh, ln);
      acc0 = wmma_bf16(afrag(ACT1, H1, mrow0 +  0, kk * 32, lh, ln), bf, acc0);
      acc1 = wmma_bf16(afrag(ACT1, H1, mrow0 + 16, kk * 32, lh, ln), bf, acc1);
      acc2 = wmma_bf16(afrag(ACT1, H1, mrow0 + 32, kk * 32, lh, ln), bf, acc2);
      acc3 = wmma_bf16(afrag(ACT1, H1, mrow0 + 48, kk * 32, lh, ln), bf, acc3);
    }
    v8f accs[4] = {acc0, acc1, acc2, acc3};
#pragma unroll
    for (int m = 0; m < 4; ++m)
#pragma unroll
      for (int v = 0; v < 8; ++v) {
        int row = mrow0 + m * 16 + v + lh * 8;
        unsafeAtomicAdd(aggr + (size_t)DSTS[row] * H2 + n0 + ln, lrelu(accs[m][v]));
      }
  }
}

// ---------------------------------------------------------------------------
// Node mm3: h = leaky([x0, aggr] @ W3 + b3)  (K = 640 -> 1024), bf16 out.
// Same 2 M-group x 4 N-group wave blocking.
__global__ void __launch_bounds__(256) node_mm3(const float* __restrict__ x0,
                                                const float* __restrict__ aggr,
                                                const __bf16* __restrict__ W3t,
                                                const float* __restrict__ b3,
                                                __bf16* __restrict__ hout) {
  extern __shared__ char smem[];
  __bf16* A1 = (__bf16*)smem; // 128 x 640 bf16 = 160 KB
  const int t = threadIdx.x, blk = blockIdx.x;
  const int K = DD + H2; // 640

  { // stage concat(x0, aggr) as bf16
    int r = t >> 1, half = t & 1;
    int rowg = blk * 128 + r;
    if (rowg >= NN) rowg = NN - 1;
    const float4* xr = (const float4*)(x0 + (size_t)rowg * DD);
    const float4* gr = (const float4*)(aggr + (size_t)rowg * H2);
    uint2* orow = (uint2*)(A1 + (size_t)r * K);
#pragma unroll 4
    for (int i = 0; i < 80; ++i) {
      int c4 = half * 80 + i;
      int col = c4 * 4;
      float4 vq = (col < DD) ? xr[c4] : gr[(col - DD) >> 2];
      uint2 pk;
      pk.x = pk2(vq.x, vq.y);
      pk.y = pk2(vq.z, vq.w);
      orow[c4] = pk;
    }
  }
  __syncthreads();

  const int wave = t >> 5, lane = t & 31;
  const int lh = lane >> 4, ln = lane & 15;
  const int mg = wave & 1, wg = wave >> 1;
  const int mrow0 = mg * 64;

  for (int i = 0; i < 16; ++i) { // 64 N-tiles / 4 N-groups
    int n0 = (wg + 4 * i) * 16;
    float bias = b3[n0 + ln];
    v8f acc0 = splat8(bias), acc1 = splat8(bias), acc2 = splat8(bias), acc3 = splat8(bias);
    for (int kk = 0; kk < K / 32; ++kk) {
      v16bf bf = bfrag(W3t, K, n0, kk * 32, lh, ln);
      acc0 = wmma_bf16(afrag(A1, K, mrow0 +  0, kk * 32, lh, ln), bf, acc0);
      acc1 = wmma_bf16(afrag(A1, K, mrow0 + 16, kk * 32, lh, ln), bf, acc1);
      acc2 = wmma_bf16(afrag(A1, K, mrow0 + 32, kk * 32, lh, ln), bf, acc2);
      acc3 = wmma_bf16(afrag(A1, K, mrow0 + 48, kk * 32, lh, ln), bf, acc3);
    }
    v8f accs[4] = {acc0, acc1, acc2, acc3};
#pragma unroll
    for (int m = 0; m < 4; ++m)
#pragma unroll
      for (int v = 0; v < 8; ++v) {
        int rowg = blk * 128 + mrow0 + m * 16 + v + lh * 8;
        if (rowg < NN) hout[(size_t)rowg * H3 + n0 + ln] = (__bf16)lrelu(accs[m][v]);
      }
  }
}

// ---------------------------------------------------------------------------
// Node mm4 + BN(eval) + residual average. A-fragments straight from global h.
__global__ void __launch_bounds__(256) node_mm4(const __bf16* __restrict__ h,
                                                const __bf16* __restrict__ W4t,
                                                const float* __restrict__ b4,
                                                const float* __restrict__ gamma,
                                                const float* __restrict__ beta,
                                                const float* __restrict__ mean,
                                                const float* __restrict__ var,
                                                const float* __restrict__ x0,
                                                float* __restrict__ out) {
  const int t = threadIdx.x, blk = blockIdx.x;
  const int wave = t >> 5, lane = t & 31;
  const int lh = lane >> 4, ln = lane & 15;
  const int mg = wave & 1, wg = wave >> 1;
  const int mrow0 = blk * 128 + mg * 64;

  // Per-M-tile A row pointers (clamped for the ragged last block)
  const __bf16* hr[4];
#pragma unroll
  for (int m = 0; m < 4; ++m) {
    int r = mrow0 + m * 16 + ln;
    if (r >= NN) r = NN - 1;
    hr[m] = h + (size_t)r * H3;
  }

  for (int i = 0; i < 2; ++i) { // 8 N-tiles / 4 N-groups
    int n0 = (wg + 4 * i) * 16;
    int col = n0 + ln;
    float bias = b4[col];
    float sc = gamma[col] * rsqrtf(var[col] + EPSC);
    float sh = beta[col] - mean[col] * sc;
    v8f acc0 = splat8(bias), acc1 = splat8(bias), acc2 = splat8(bias), acc3 = splat8(bias);
    for (int kk = 0; kk < H3 / 32; ++kk) {
      v16bf bf = bfrag(W4t, H3, n0, kk * 32, lh, ln);
      const int off = kk * 32 + lh * 8;
      {
        const char* p = (const char*)(hr[0] + off);
        acc0 = wmma_bf16(make_frag(*(const uint4*)p, *(const uint4*)(p + 32)), bf, acc0);
      }
      {
        const char* p = (const char*)(hr[1] + off);
        acc1 = wmma_bf16(make_frag(*(const uint4*)p, *(const uint4*)(p + 32)), bf, acc1);
      }
      {
        const char* p = (const char*)(hr[2] + off);
        acc2 = wmma_bf16(make_frag(*(const uint4*)p, *(const uint4*)(p + 32)), bf, acc2);
      }
      {
        const char* p = (const char*)(hr[3] + off);
        acc3 = wmma_bf16(make_frag(*(const uint4*)p, *(const uint4*)(p + 32)), bf, acc3);
      }
    }
    v8f accs[4] = {acc0, acc1, acc2, acc3};
#pragma unroll
    for (int m = 0; m < 4; ++m)
#pragma unroll
      for (int v = 0; v < 8; ++v) {
        int rowg = mrow0 + m * 16 + v + lh * 8;
        if (rowg < NN) {
          float val = accs[m][v] * sc + sh; // bias already in acc
          out[(size_t)rowg * DD + col] = (val + x0[(size_t)rowg * DD + col]) * 0.5f;
        }
      }
  }
}

// ---------------------------------------------------------------------------
extern "C" void kernel_launch(void* const* d_in, const int* in_sizes, int n_in,
                              void* d_out, int out_size, void* d_ws, size_t ws_size,
                              hipStream_t stream) {
  (void)in_sizes; (void)n_in; (void)out_size;
  const float* x0    = (const float*)d_in[0];
  const int*   eidx  = (const int*)d_in[1];
  const float* eattr = (const float*)d_in[2];
  const float* W1 = (const float*)d_in[3];  const float* b1 = (const float*)d_in[4];
  const float* W2 = (const float*)d_in[5];  const float* b2 = (const float*)d_in[6];
  const float* W3 = (const float*)d_in[7];  const float* b3 = (const float*)d_in[8];
  const float* W4 = (const float*)d_in[9];  const float* b4 = (const float*)d_in[10];
  const float* gm = (const float*)d_in[11]; const float* bt = (const float*)d_in[12];
  const float* mn = (const float*)d_in[13]; const float* vr = (const float*)d_in[14];
  float* out = (float*)d_out;

  // Workspace layout (all offsets 256B-aligned)
  const size_t AGGR_OFF = 0;                            // 20000*512*4  = 40,960,000
  const size_t H_OFF    = 40960000;                     // 20000*1024*2 = 40,960,000
  const size_t W1T_OFF  = 81920000;                     // 256*128*2
  const size_t W2T_OFF  = 81985536;                     // 512*256*2
  const size_t W3T_OFF  = 82247680;                     // 1024*640*2
  const size_t W4T_OFF  = 83558400;                     // 128*1024*2
  const size_t WS_NEED  = 83820544;
  if (ws_size < WS_NEED) return;

  char* ws = (char*)d_ws;
  float*  aggr = (float*)(ws + AGGR_OFF);
  __bf16* hbuf = (__bf16*)(ws + H_OFF);
  __bf16* W1t  = (__bf16*)(ws + W1T_OFF);
  __bf16* W2t  = (__bf16*)(ws + W2T_OFF);
  __bf16* W3t  = (__bf16*)(ws + W3T_OFF);
  __bf16* W4t  = (__bf16*)(ws + W4T_OFF);

  hipMemsetAsync(aggr, 0, (size_t)NN * H2 * sizeof(float), stream);

  cvt_transpose<<<(DD * H1 + 255) / 256, 256, 0, stream>>>(W1, W1t, DD, H1);
  cvt_transpose<<<(H1 * H2 + 255) / 256, 256, 0, stream>>>(W2, W2t, H1, H2);
  cvt_transpose<<<((DD + H2) * H3 + 255) / 256, 256, 0, stream>>>(W3, W3t, DD + H2, H3);
  cvt_transpose<<<(H3 * DD + 255) / 256, 256, 0, stream>>>(W4, W4t, H3, DD);

  const size_t smem_edge = 128 * DD * 2 + 128 * H1 * 2 + 128 * 4;  // 98,816 B
  const size_t smem_node = 128 * (DD + H2) * 2;                    // 163,840 B
  hipFuncSetAttribute(reinterpret_cast<const void*>(edge_mlp),
                      hipFuncAttributeMaxDynamicSharedMemorySize, (int)smem_edge);
  hipFuncSetAttribute(reinterpret_cast<const void*>(node_mm3),
                      hipFuncAttributeMaxDynamicSharedMemorySize, (int)smem_node);

  edge_mlp<<<EE / 128, 256, smem_edge, stream>>>(x0, eidx, eattr, W1t, b1, W2t, b2, aggr);

  const int nb = (NN + 127) / 128;
  node_mm3<<<nb, 256, smem_node, stream>>>(x0, aggr, W3t, b3, hbuf);
  node_mm4<<<nb, 256, 0, stream>>>(hbuf, W4t, b4, gm, bt, mn, vr, x0, out);
}
